// MSARowWiseAttentionWithPairBias_36060545417652
// MI455X (gfx1250) — matrix operand order repaired and checked by reading, not compiled, not run on hardware
//
#include <hip/hip_runtime.h>
#include <hip/hip_bf16.h>

typedef __attribute__((ext_vector_type(16))) _Float16 v16h;
typedef __attribute__((ext_vector_type(8)))  _Float16 v8h;
typedef __attribute__((ext_vector_type(8)))  float    v8f;
typedef __attribute__((ext_vector_type(4)))  float    v4f;

#define XOR_SWZ(v, M) __int_as_float(__builtin_amdgcn_ds_swizzle(__float_as_int(v), ((M) << 10) | 0x1f))

// ---- problem constants ----
#define BS   128        // b*s rows of MSA
#define RR   384        // residues
#define CM   256        // c_msa = H*D
#define CP   128        // c_pair
#define NH   8
#define DD   32
#define MROWS (BS*RR)   // 49152

// ---------------- WMMA fragment helpers ----------------
// A fragment (16x32 f16), source row-major [M][K] with leading dim ld (halves).
__device__ __forceinline__ v16h ldA16(const _Float16* base, int ld) {
  int lane = threadIdx.x & 31;
  const _Float16* p = base + (lane & 15) * ld + ((lane >> 4) << 3);
  v8h lo = *(const v8h*)p;        // k = 8g + 0..7
  v8h hi = *(const v8h*)(p + 16); // k = 8g + 16..23
  v16h a;
#pragma unroll
  for (int i = 0; i < 8; ++i) { a[i] = lo[i]; a[i + 8] = hi[i]; }
  return a;
}

// A fragment from f32 source (converted to f16).
__device__ __forceinline__ v16h ldA16_f32(const float* base, int ld) {
  int lane = threadIdx.x & 31;
  const float* p = base + (lane & 15) * ld + ((lane >> 4) << 3);
  v16h a;
#pragma unroll
  for (int i = 0; i < 8; ++i) {
    a[i]     = (_Float16)p[i];
    a[i + 8] = (_Float16)p[i + 16];
  }
  return a;
}

// B fragment (32x16 f16) from transposed storage Bt[N][K], leading dim ld (halves).
// lane = column N, k = 8*(lane>>4) + i and 16 + 8*(lane>>4) + i (same as A).
__device__ __forceinline__ v16h ldBt16(const _Float16* base, int ld) {
  int lane = threadIdx.x & 31;
  const _Float16* p = base + (lane & 15) * ld + ((lane >> 4) << 3);
  v8h lo = *(const v8h*)p;
  v8h hi = *(const v8h*)(p + 16);
  v16h b;
#pragma unroll
  for (int i = 0; i < 8; ++i) { b[i] = lo[i]; b[i + 8] = hi[i]; }
  return b;
}

__device__ __forceinline__ v8f wmma_f16(v16h a, v16h b, v8f c) {
  return __builtin_amdgcn_wmma_f32_16x16x32_f16(false, a, false, b, (short)0, c, false, false);
}

// ---------------- kernel 0: weight prep (f32 [K][N] -> f16 [N][K]) ----------------
__global__ void k_wprep(const float* Wq, const float* Wk, const float* Wv,
                        const float* Wg, const float* Wo, const float* Wb,
                        _Float16* wqt, _Float16* wkt, _Float16* wvt,
                        _Float16* wgt, _Float16* wot, _Float16* wbt) {
  int idx = blockIdx.x * blockDim.x + threadIdx.x;
  if (idx < 5 * 65536) {
    int wsel = idx >> 16, e = idx & 65535;
    int n = e >> 8, k = e & 255;
    const float* src = wsel == 0 ? Wq : wsel == 1 ? Wk : wsel == 2 ? Wv : wsel == 3 ? Wg : Wo;
    _Float16* dst    = wsel == 0 ? wqt : wsel == 1 ? wkt : wsel == 2 ? wvt : wsel == 3 ? wgt : wot;
    dst[n * 256 + k] = (_Float16)src[k * 256 + n];
  } else if (idx < 5 * 65536 + 2048) {
    int e = idx - 5 * 65536;
    int n = e >> 7, k = e & 127;                   // pad N: 8 -> 16 with zeros
    wbt[n * 128 + k] = (n < NH) ? (_Float16)Wb[k * NH + n] : (_Float16)0.f;
  }
}

// ---------------- kernel 1: LayerNorm over residue axis r ----------------
__global__ void k_ln(const float* msa, const float* lng, const float* lnb, _Float16* xh) {
  int s = blockIdx.x;          // 0..127
  int c = threadIdx.x;         // 0..255
  const float* base = msa + (size_t)s * RR * CM + c;
  float sum = 0.f, sq = 0.f;
  for (int r = 0; r < RR; ++r) {
    float v = base[r * CM];
    sum += v; sq += v * v;
  }
  float mean = sum * (1.f / RR);
  float var  = sq * (1.f / RR) - mean * mean;
  float rs   = rsqrtf(var + 1e-5f);
  _Float16* ob = xh + (size_t)s * RR * CM + c;
  for (int r = 0; r < RR; ++r) {
    float v = (base[r * CM] - mean) * rs * lng[r] + lnb[r];
    ob[r * CM] = (_Float16)v;
  }
}

// ---------------- kernel 2: pair bias = pair @ Wb + bb  (WMMA, N padded to 16) ----------------
__global__ void k_bias(const float* pair, const _Float16* wbt, const float* bbv, float* biasb) {
  int mtile = blockIdx.x * (blockDim.x >> 5) + (threadIdx.x >> 5); // 0..9215
  const float* ab = pair + (size_t)mtile * 16 * CP;
  v8f c = {};
#pragma unroll
  for (int kt = 0; kt < 4; ++kt) {
    v16h a = ldA16_f32(ab + kt * 32, CP);
    v16h b = ldBt16(wbt + kt * 32, CP);
    c = wmma_f16(a, b, c);
  }
  int lane = threadIdx.x & 31;
  int n = lane & 15;
  int row0 = mtile * 16 + ((lane >> 4) << 3);
  if (n < NH) {
    float bv = bbv[n];
    v4f lo, hi;
#pragma unroll
    for (int j = 0; j < 4; ++j) { lo[j] = c[j] + bv; hi[j] = c[j + 4] + bv; }
    float* dst = biasb + (size_t)n * (RR * RR) + row0;   // bias[h][i*384+j], 8 contiguous
    *(v4f*)dst = lo;
    *(v4f*)(dst + 4) = hi;
  }
}

// ---------------- kernel 3: fused Q/K/V/G projections (16x64 per wave) ----------------
__global__ void k_qkvg(const _Float16* xh, const _Float16* wqt, const _Float16* wkt,
                       const _Float16* wvt, const _Float16* wgt, const float* bg,
                       _Float16* qh, _Float16* kh, _Float16* vt, _Float16* gh) {
  int wid = blockIdx.x * (blockDim.x >> 5) + (threadIdx.x >> 5);
  int ng     = wid & 3;                         // group of 4 ntiles
  int weight = (wid >> 2) & 3;
  int mtile  = wid >> 4;                        // 0..3071
  const _Float16* wt = weight == 0 ? wqt : weight == 1 ? wkt : weight == 2 ? wvt : wgt;
  const _Float16* abase = xh + (size_t)mtile * 16 * CM;
  const _Float16* bbase = wt + (size_t)ng * 64 * CM;
  v8f acc[4] = {{}, {}, {}, {}};
#pragma unroll
  for (int kt = 0; kt < 8; ++kt) {
    v16h a = ldA16(abase + kt * 32, CM);        // loaded once, used by 4 WMMAs
#pragma unroll
    for (int u = 0; u < 4; ++u)
      acc[u] = wmma_f16(a, ldBt16(bbase + (size_t)u * 16 * CM + kt * 32, CM), acc[u]);
  }

  int lane = threadIdx.x & 31;
  int row0 = mtile * 16 + ((lane >> 4) << 3);   // rows of a tile stay in one bs
  int bs = row0 / RR;
  int r0 = row0 - bs * RR;
#pragma unroll
  for (int u = 0; u < 4; ++u) {
    int col = (ng * 4 + u) * 16 + (lane & 15);  // 0..255
    int h = col >> 5, d = col & 31;
    if (weight == 3) {
      float bgv = bg[col];
#pragma unroll
      for (int j = 0; j < 8; ++j) {
        float gv = 1.0f / (1.0f + __expf(-(acc[u][j] + bgv)));
        gh[(size_t)(row0 + j) * CM + col] = (_Float16)gv;
      }
    } else if (weight == 2) {                   // V transposed: [bsh][d][r], 8 contiguous r
      v8h pk;
#pragma unroll
      for (int j = 0; j < 8; ++j) pk[j] = (_Float16)acc[u][j];
      *(v8h*)(vt + ((size_t)(bs * NH + h) * DD + d) * RR + r0) = pk;
    } else {                                    // Q/K: [bsh][r][d]
      _Float16* dst = (weight == 0) ? qh : kh;
#pragma unroll
      for (int j = 0; j < 8; ++j)
        dst[((size_t)(bs * NH + h) * RR + (r0 + j)) * DD + d] = (_Float16)acc[u][j];
    }
  }
}

// ---------------- kernel 4: flash attention with pair bias + gate ----------------
__global__ void k_attn(const _Float16* qh, const _Float16* kh, const _Float16* vt,
                       const float* biasb, const _Float16* gh, _Float16* oh) {
  __shared__ __align__(32) _Float16 pbuf[8][16 * 32];  // per-wave 16x32 prob tile
  int w    = threadIdx.x >> 5;
  int lane = threadIdx.x & 31;
  int bsh    = blockIdx.x / 3;                  // 0..1023
  int qchunk = blockIdx.x % 3;
  int bs = bsh >> 3, h = bsh & 7;
  int qbase = qchunk * 128 + w * 16;            // query row base in [0,384)
  const _Float16* qp = qh + ((size_t)bsh * RR + qbase) * DD;
  const _Float16* kp = kh + (size_t)bsh * RR * DD;
  const _Float16* vp = vt + (size_t)bsh * DD * RR;
  const float*    bp = biasb + (size_t)h * RR * RR;

  v16h aQ = ldA16(qp, DD);                      // full D=32 in one fragment
  float mrun[8], lrun[8];
#pragma unroll
  for (int j = 0; j < 8; ++j) { mrun[j] = -1e30f; lrun[j] = 0.f; }
  v8f acc0 = {}, acc1 = {};
  int colL = lane & 15;
  int rowg = (lane >> 4) << 3;
  const float scale = 0.17677669529663687f;     // 1/sqrt(32)

  for (int t = 0; t < 12; ++t) {                // 12 steps of 32 keys
    int jc0 = t * 32, jc1 = t * 32 + 16;
    if (t < 11) {                               // prefetch next K/V tiles (global_prefetch_b8)
      __builtin_prefetch(kp + (size_t)(jc0 + 32) * DD, 0, 1);
      __builtin_prefetch(vp + t * 32 + 32, 0, 1);
      __builtin_prefetch(vp + 16 * RR + t * 32 + 32, 0, 1);
    }
    v8f z = {};
    v8f d0 = wmma_f16(aQ, ldBt16(kp + (size_t)jc0 * DD, DD), z);
    v8f d1 = wmma_f16(aQ, ldBt16(kp + (size_t)jc1 * DD, DD), z);
#pragma unroll
    for (int j = 0; j < 8; ++j) {
      int row = qbase + rowg + j;
      float s0 = d0[j] * scale + bp[(size_t)row * RR + jc0 + colL];
      float s1 = d1[j] * scale + bp[(size_t)row * RR + jc1 + colL];
      float tm = fmaxf(s0, s1);                 // row max across 16 lanes (xor stays in half)
      tm = fmaxf(tm, XOR_SWZ(tm, 1));
      tm = fmaxf(tm, XOR_SWZ(tm, 2));
      tm = fmaxf(tm, XOR_SWZ(tm, 4));
      tm = fmaxf(tm, XOR_SWZ(tm, 8));
      float nm = fmaxf(mrun[j], tm);
      float cf = __expf(mrun[j] - nm);
      mrun[j] = nm;
      float p0 = __expf(s0 - nm);
      float p1 = __expf(s1 - nm);
      float rs = p0 + p1;
      rs += XOR_SWZ(rs, 1);
      rs += XOR_SWZ(rs, 2);
      rs += XOR_SWZ(rs, 4);
      rs += XOR_SWZ(rs, 8);
      lrun[j] = lrun[j] * cf + rs;
      acc0[j] *= cf; acc1[j] *= cf;
      int rl = rowg + j;                        // C layout -> A layout via LDS
      pbuf[w][rl * 32 + colL]      = (_Float16)p0;
      pbuf[w][rl * 32 + 16 + colL] = (_Float16)p1;
    }
    __syncthreads();
    v16h aP = ldA16(&pbuf[w][0], 32);
    acc0 = wmma_f16(aP, ldBt16(vp + t * 32, RR), acc0);            // d = 0..15
    acc1 = wmma_f16(aP, ldBt16(vp + 16 * RR + t * 32, RR), acc1);  // d = 16..31
    __syncthreads();
  }
  // epilogue: normalize, gate, scatter to [m][h*32+d]
#pragma unroll
  for (int j = 0; j < 8; ++j) {
    int r = qbase + rowg + j;
    size_t m = (size_t)bs * RR + r;
    float inv = 1.0f / lrun[j];
    int c0 = h * 32 + colL;
    int c1 = c0 + 16;
    float g0 = (float)gh[m * CM + c0];
    float g1 = (float)gh[m * CM + c1];
    oh[m * CM + c0] = (_Float16)(acc0[j] * inv * g0);
    oh[m * CM + c1] = (_Float16)(acc1[j] * inv * g1);
  }
}

// ---------------- kernel 5: output projection + bo (16x64 per wave) ----------------
__global__ void k_outproj(const _Float16* oh, const _Float16* wot, const float* bo, float* out) {
  int wid = blockIdx.x * (blockDim.x >> 5) + (threadIdx.x >> 5);
  int ng    = wid & 3;                           // group of 4 ntiles
  int mtile = wid >> 2;                          // 0..3071
  const _Float16* ab  = oh + (size_t)mtile * 16 * CM;
  const _Float16* bbw = wot + (size_t)ng * 64 * CM;
  v8f acc[4] = {{}, {}, {}, {}};
#pragma unroll
  for (int kt = 0; kt < 8; ++kt) {
    v16h a = ldA16(ab + kt * 32, CM);
#pragma unroll
    for (int u = 0; u < 4; ++u)
      acc[u] = wmma_f16(a, ldBt16(bbw + (size_t)u * 16 * CM + kt * 32, CM), acc[u]);
  }
  int lane = threadIdx.x & 31;
  int row0 = mtile * 16 + ((lane >> 4) << 3);
#pragma unroll
  for (int u = 0; u < 4; ++u) {
    int col = (ng * 4 + u) * 16 + (lane & 15);
    float bov = bo[col];
#pragma unroll
    for (int j = 0; j < 8; ++j)
      out[(size_t)(row0 + j) * CM + col] = acc[u][j] + bov;
  }
}

// ---------------- launch ----------------
extern "C" void kernel_launch(void* const* d_in, const int* in_sizes, int n_in,
                              void* d_out, int out_size, void* d_ws, size_t ws_size,
                              hipStream_t stream) {
  (void)in_sizes; (void)n_in; (void)out_size; (void)ws_size;
  const float* msa  = (const float*)d_in[0];
  const float* pair = (const float*)d_in[1];
  const float* lng  = (const float*)d_in[2];
  const float* lnb  = (const float*)d_in[3];
  const float* Wq   = (const float*)d_in[4];
  const float* Wk   = (const float*)d_in[5];
  const float* Wv   = (const float*)d_in[6];
  const float* Wb   = (const float*)d_in[7];
  const float* bb   = (const float*)d_in[8];
  const float* Wg   = (const float*)d_in[9];
  const float* bg   = (const float*)d_in[10];
  const float* Wo   = (const float*)d_in[11];
  const float* bo   = (const float*)d_in[12];

  char* ws = (char*)d_ws;
  size_t off = 0;
  auto take = [&](size_t bytes) { char* p = ws + off; off += (bytes + 255) & ~(size_t)255; return p; };
  const size_t SZ_ACT = (size_t)MROWS * CM * 2;       // 25.17 MB each
  _Float16* xh  = (_Float16*)take(SZ_ACT);            // also reused as oh after k_qkvg
  _Float16* qh  = (_Float16*)take(SZ_ACT);
  _Float16* kh  = (_Float16*)take(SZ_ACT);
  _Float16* vt  = (_Float16*)take(SZ_ACT);
  _Float16* gh  = (_Float16*)take(SZ_ACT);
  float*    biasb = (float*)take((size_t)NH * RR * RR * 4);
  _Float16* wqt = (_Float16*)take(65536 * 2);
  _Float16* wkt = (_Float16*)take(65536 * 2);
  _Float16* wvt = (_Float16*)take(65536 * 2);
  _Float16* wgt = (_Float16*)take(65536 * 2);
  _Float16* wot = (_Float16*)take(65536 * 2);
  _Float16* wbt = (_Float16*)take(2048 * 2);
  _Float16* oh = xh;                                  // xh dead after k_qkvg

  k_wprep<<<(5 * 65536 + 2048 + 255) / 256, 256, 0, stream>>>(
      Wq, Wk, Wv, Wg, Wo, Wb, wqt, wkt, wvt, wgt, wot, wbt);
  k_ln<<<BS, 256, 0, stream>>>(msa, lng, lnb, xh);
  k_bias<<<(RR * RR / 16) / 8, 256, 0, stream>>>(pair, wbt, bb, biasb);
  k_qkvg<<<(MROWS / 16) * 4 * 4 / 8, 256, 0, stream>>>(
      xh, wqt, wkt, wvt, wgt, bg, qh, kh, vt, gh);
  k_attn<<<BS * NH * 3, 256, 0, stream>>>(qh, kh, vt, biasb, gh, oh);
  k_outproj<<<(MROWS / 16) * 4 / 8, 256, 0, stream>>>(oh, wot, bo, (float*)d_out);
}
